// Attention_38577396252867
// MI455X (gfx1250) — matrix operand order repaired and checked by reading, not compiled, not run on hardware
//
#include <hip/hip_runtime.h>
#include <hip/hip_bf16.h>

// ---- problem constants (from reference) ----
constexpr int B_ = 8;
constexpr int N_ = 2048;
constexpr int C_ = 384;
constexpr int H_ = 6;
constexpr int D_ = 64;
constexpr float SCALE_ = 0.125f; // 1/sqrt(64)

typedef __attribute__((ext_vector_type(16))) _Float16 v16h;
typedef __attribute__((ext_vector_type(8)))  _Float16 v8h;
typedef __attribute__((ext_vector_type(8)))  float    v8f;

// D = A(16x32 f16) x B(32x16 f16) + C(16x16 f32)
__device__ inline v8f wmma_f16(v16h a, v16h b, v8f c) {
  return __builtin_amdgcn_wmma_f32_16x16x32_f16(
      /*neg_a=*/false, a, /*neg_b=*/false, b,
      /*c_mod=*/(short)0, c, /*reuse_a=*/false, /*reuse_b=*/false);
}

// A fragment (16x32, f16) from row-major src. `base` points at row0 of the tile.
// ISA layout: lanes 0-15 hold rows M=lane, elems 0..7 = K[k0+hs*8 .. +7],
// elems 8..15 = K[k0+hs*8+16 .. +23] (hs = lane>>4).
__device__ inline v16h load_frag_a(const _Float16* base, int ld, int k0) {
  const int lane = threadIdx.x & 31;
  const _Float16* p = base + (size_t)(lane & 15) * ld + k0 + (lane >> 4) * 8;
  v8h lo = *(const v8h*)p;
  v8h hi = *(const v8h*)(p + 16);
  v16h r;
#pragma unroll
  for (int i = 0; i < 8; ++i) { r[i] = lo[i]; r[i + 8] = hi[i]; }
  return r;
}

// B fragment (32x16, f16) where B[k][n] = w[n][k] (w row-major, leading dim ld).
// Lane holds column n = n0 + (lane&15); elems 0..7 = K[k0+hs*8 .. +7],
// elems 8..15 = K[k0+hs*8+16 .. +23] (matching A). `base` points at row n0.
__device__ inline v16h load_frag_bT(const _Float16* base, int ld, int k0) {
  const int lane = threadIdx.x & 31;
  const _Float16* p = base + (size_t)(lane & 15) * ld + k0 + (lane >> 4) * 8;
  v8h lo = *(const v8h*)p;
  v8h hi = *(const v8h*)(p + 16);
  v16h r;
#pragma unroll
  for (int i = 0; i < 8; ++i) { r[i] = lo[i]; r[i + 8] = hi[i]; }
  return r;
}

// A fragment from a 16x32 row-major f16 tile in LDS.
__device__ inline v16h load_frag_a_lds(const _Float16* base) {
  const int lane = threadIdx.x & 31;
  const _Float16* p = base + (lane & 15) * 32 + (lane >> 4) * 8;
  v16h r;
#pragma unroll
  for (int i = 0; i < 8; ++i) { r[i] = p[i]; r[i + 8] = p[16 + i]; }
  return r;
}

// ---------------- stage 0: f32 -> f16 convert ----------------
__global__ void f32_to_f16_kernel(const float* __restrict__ src,
                                  _Float16* __restrict__ dst, int n) {
  int i = blockIdx.x * blockDim.x + threadIdx.x;
  int stride = gridDim.x * blockDim.x;
  for (; i < n; i += stride) dst[i] = (_Float16)src[i];
}

// ---------------- stage 1: fused QKV projection (2x2 blocked) ----------------
// QKV[m, col] = sum_k x[m,k] * w_qkv[col,k];  col = t*C + h*D + d
// Wave computes a 32x32 output block (4 WMMA tiles) -> 4 WMMAs per 4 frag loads.
// Scatter into Q[B,H,N,D], K[B,H,N,D], V^T[B,H,D,N] (all f16).
__global__ void __launch_bounds__(256) qkv_gemm_kernel(
    const _Float16* __restrict__ x16, const _Float16* __restrict__ w16,
    _Float16* __restrict__ q16, _Float16* __restrict__ k16,
    _Float16* __restrict__ vt16) {
  const int wave = threadIdx.x >> 5;
  const int lane = threadIdx.x & 31;
  const int tile = blockIdx.x * 8 + wave;
  const int ntn = (3 * C_) / 32;            // 36 col blocks of 32
  const int m0 = (tile / ntn) * 32;
  const int n0 = (tile % ntn) * 32;

  const _Float16* a0base = x16 + (size_t)m0 * C_;
  const _Float16* a1base = a0base + (size_t)16 * C_;
  const _Float16* b0base = w16 + (size_t)n0 * C_;
  const _Float16* b1base = b0base + (size_t)16 * C_;
  v8f c00 = {}, c01 = {}, c10 = {}, c11 = {};
#pragma unroll 2
  for (int k0 = 0; k0 < C_; k0 += 32) {
    v16h a0 = load_frag_a(a0base, C_, k0);
    v16h a1 = load_frag_a(a1base, C_, k0);
    v16h b0 = load_frag_bT(b0base, C_, k0);
    v16h b1 = load_frag_bT(b1base, C_, k0);
    c00 = wmma_f16(a0, b0, c00);
    c01 = wmma_f16(a0, b1, c01);
    c10 = wmma_f16(a1, b0, c10);
    c11 = wmma_f16(a1, b1, c11);
  }

  const int hs = lane >> 4;
  const int ln = lane & 15;
#pragma unroll
  for (int sn = 0; sn < 2; ++sn) {
    const int ncol = n0 + sn * 16 + ln;
    const int t = ncol / C_;
    const int rem = ncol % C_;
    const int h = rem / D_;
    const int d = rem % D_;
#pragma unroll
    for (int sm = 0; sm < 2; ++sm) {
      const v8f& c = sn == 0 ? (sm == 0 ? c00 : c10) : (sm == 0 ? c01 : c11);
#pragma unroll
      for (int r = 0; r < 8; ++r) {
        const int m = m0 + sm * 16 + r + 8 * hs;
        const int bb = m / N_;
        const int row = m % N_;
        const size_t bh = (size_t)bb * H_ + h;
        const _Float16 v = (_Float16)c[r];
        if (t == 0)      q16[(bh * N_ + row) * D_ + d] = v;
        else if (t == 1) k16[(bh * N_ + row) * D_ + d] = v;
        else             vt16[(bh * D_ + d) * N_ + row] = v;
      }
    }
  }
}

// ---------------- stage 2: attention core ----------------
// One wave owns a 16-row strip x all N cols of one (b,h).
// Pass1: S = scale*Q K^T (WMMA) -> raw f32 scores to attn, track row max.
// Pass2: re-read strip (L2 hot), exp, rowsum, P*V via WMMA (P transposed
//        through per-wave LDS tile into A-fragment layout). Write O (f16).
// Pass3: re-read strip, write normalized attn.
__global__ void __launch_bounds__(256) attention_kernel(
    const _Float16* __restrict__ q16, const _Float16* __restrict__ k16,
    const _Float16* __restrict__ vt16, _Float16* __restrict__ o16,
    float* __restrict__ attn) {
  __shared__ _Float16 ptile[8][16 * 32];    // per-wave 16x32 staging (8 KB)

  const int wave = threadIdx.x >> 5;
  const int lane = threadIdx.x & 31;
  const int hs = lane >> 4;
  const int ln = lane & 15;

  const int bh = blockIdx.x >> 4;                    // 0..47
  const int strip = ((blockIdx.x & 15) << 3) + wave; // 0..127
  const int row0 = strip * 16;
  const int bb = bh / H_;
  const int h = bh % H_;

  const _Float16* qb = q16 + (size_t)bh * N_ * D_;
  const _Float16* kb = k16 + (size_t)bh * N_ * D_;
  const _Float16* vb = vt16 + (size_t)bh * D_ * N_;
  float* arow = attn + ((size_t)bh * N_ + row0) * N_;

  const v16h aq0 = load_frag_a(qb + (size_t)row0 * D_, D_, 0);
  const v16h aq1 = load_frag_a(qb + (size_t)row0 * D_, D_, 32);

  float vmax[8];
#pragma unroll
  for (int r = 0; r < 8; ++r) vmax[r] = -3.0e38f;

  // ---- pass 1: scores ----
  for (int ct = 0; ct < N_ / 16; ++ct) {
    const int col0 = ct * 16;
    v16h b0 = load_frag_bT(kb + (size_t)col0 * D_, D_, 0);
    v16h b1 = load_frag_bT(kb + (size_t)col0 * D_, D_, 32);
    v8f c = {};
    c = wmma_f16(aq0, b0, c);
    c = wmma_f16(aq1, b1, c);
#pragma unroll
    for (int r = 0; r < 8; ++r) {
      const float s = c[r] * SCALE_;
      vmax[r] = fmaxf(vmax[r], s);
      arow[(size_t)(r + 8 * hs) * N_ + col0 + ln] = s;
    }
  }
  // row-max reduce across the 16 lanes sharing each row (xor masks stay in half)
#pragma unroll
  for (int mm = 1; mm < 16; mm <<= 1)
#pragma unroll
    for (int r = 0; r < 8; ++r)
      vmax[r] = fmaxf(vmax[r], __shfl_xor(vmax[r], mm, 32));

  // ---- pass 2: exp + rowsum + O = P*V ----
  float rowsum[8];
  v8f co[4];
  {
    v8f z = {};
#pragma unroll
    for (int nt = 0; nt < 4; ++nt) co[nt] = z;
#pragma unroll
    for (int r = 0; r < 8; ++r) rowsum[r] = 0.f;
  }
  _Float16* myp = &ptile[wave][0];

  for (int ck = 0; ck < N_ / 32; ++ck) {
    const int col0 = ck * 32;
#pragma unroll
    for (int r = 0; r < 8; ++r) {
      const size_t base = (size_t)(r + 8 * hs) * N_ + col0 + ln;
      const float p0 = __expf(arow[base] - vmax[r]);
      const float p1 = __expf(arow[base + 16] - vmax[r]);
      rowsum[r] += p0 + p1;
      myp[(r + 8 * hs) * 32 + ln] = (_Float16)p0;
      myp[(r + 8 * hs) * 32 + 16 + ln] = (_Float16)p1;
    }
    __syncthreads();                       // cross-lane LDS visibility
    const v16h ap = load_frag_a_lds(myp);
    __syncthreads();                       // protect tile before next overwrite
#pragma unroll
    for (int nt = 0; nt < 4; ++nt) {
      v16h bv = load_frag_bT(vb + (size_t)(nt * 16) * N_, N_, col0);
      co[nt] = wmma_f16(ap, bv, co[nt]);
    }
  }

#pragma unroll
  for (int mm = 1; mm < 16; mm <<= 1)
#pragma unroll
    for (int r = 0; r < 8; ++r)
      rowsum[r] += __shfl_xor(rowsum[r], mm, 32);

  float inv[8];
#pragma unroll
  for (int r = 0; r < 8; ++r) inv[r] = 1.0f / rowsum[r];

  // O in [B, N, C] f16 (head-interleaved, matching transpose+reshape)
#pragma unroll
  for (int nt = 0; nt < 4; ++nt)
#pragma unroll
    for (int r = 0; r < 8; ++r) {
      const int row = row0 + r + 8 * hs;
      o16[((size_t)bb * N_ + row) * C_ + h * D_ + nt * 16 + ln] =
          (_Float16)(co[nt][r] * inv[r]);
    }

  // ---- pass 3: normalized attn write ----
  for (int ct = 0; ct < N_ / 16; ++ct) {
#pragma unroll
    for (int r = 0; r < 8; ++r) {
      const size_t idx = (size_t)(r + 8 * hs) * N_ + ct * 16 + ln;
      arow[idx] = __expf(arow[idx] - vmax[r]) * inv[r];
    }
  }
}

// ---------------- stage 3: output projection (2x2 blocked) ----------------
// Y[m,n] = sum_k O[m,k] * w_proj[n,k] + b_proj[n]   (f32 result to d_out)
__global__ void __launch_bounds__(256) proj_gemm_kernel(
    const _Float16* __restrict__ o16, const _Float16* __restrict__ wp16,
    const float* __restrict__ bias, float* __restrict__ out) {
  const int wave = threadIdx.x >> 5;
  const int lane = threadIdx.x & 31;
  const int tile = blockIdx.x * 8 + wave;
  const int ntn = C_ / 32; // 12 col blocks of 32
  const int m0 = (tile / ntn) * 32;
  const int n0 = (tile % ntn) * 32;

  const _Float16* a0base = o16 + (size_t)m0 * C_;
  const _Float16* a1base = a0base + (size_t)16 * C_;
  const _Float16* b0base = wp16 + (size_t)n0 * C_;
  const _Float16* b1base = b0base + (size_t)16 * C_;
  v8f c00 = {}, c01 = {}, c10 = {}, c11 = {};
#pragma unroll 2
  for (int k0 = 0; k0 < C_; k0 += 32) {
    v16h a0 = load_frag_a(a0base, C_, k0);
    v16h a1 = load_frag_a(a1base, C_, k0);
    v16h b0 = load_frag_bT(b0base, C_, k0);
    v16h b1 = load_frag_bT(b1base, C_, k0);
    c00 = wmma_f16(a0, b0, c00);
    c01 = wmma_f16(a0, b1, c01);
    c10 = wmma_f16(a1, b0, c10);
    c11 = wmma_f16(a1, b1, c11);
  }

  const int hs = lane >> 4;
  const int ln = lane & 15;
  const float bv0 = bias[n0 + ln];
  const float bv1 = bias[n0 + 16 + ln];
#pragma unroll
  for (int r = 0; r < 8; ++r) {
    const size_t row0 = (size_t)(m0 + r + 8 * hs) * C_;
    const size_t row1 = (size_t)(m0 + 16 + r + 8 * hs) * C_;
    out[row0 + n0 + ln]      = c00[r] + bv0;
    out[row0 + n0 + 16 + ln] = c01[r] + bv1;
    out[row1 + n0 + ln]      = c10[r] + bv0;
    out[row1 + n0 + 16 + ln] = c11[r] + bv1;
  }
}

extern "C" void kernel_launch(void* const* d_in, const int* in_sizes, int n_in,
                              void* d_out, int out_size, void* d_ws, size_t ws_size,
                              hipStream_t stream) {
  (void)in_sizes; (void)n_in; (void)out_size; (void)ws_size;
  const float* x      = (const float*)d_in[0];
  const float* w_qkv  = (const float*)d_in[1];
  const float* w_proj = (const float*)d_in[2];
  const float* b_proj = (const float*)d_in[3];

  float* out  = (float*)d_out;
  float* attn = out + (size_t)B_ * N_ * C_;

  // workspace carve-up (all f16, 256B aligned; ~64 MB total)
  char* ws = (char*)d_ws;
  size_t off = 0;
  auto carve = [&](size_t elems) {
    void* p = ws + off;
    off += (elems * sizeof(_Float16) + 255) & ~(size_t)255;
    return (_Float16*)p;
  };
  const size_t XN = (size_t)B_ * N_ * C_;      // 6,291,456
  const size_t QN = (size_t)B_ * H_ * N_ * D_; // 6,291,456
  _Float16* x16     = carve(XN);
  _Float16* wqkv16  = carve((size_t)3 * C_ * C_);
  _Float16* wproj16 = carve((size_t)C_ * C_);
  _Float16* q16     = carve(QN);
  _Float16* k16     = carve(QN);
  _Float16* vt16    = carve(QN);
  _Float16* o16     = carve(XN);

  f32_to_f16_kernel<<<2048, 256, 0, stream>>>(x, x16, (int)XN);
  f32_to_f16_kernel<<<432, 256, 0, stream>>>(w_qkv, wqkv16, 3 * C_ * C_);
  f32_to_f16_kernel<<<144, 256, 0, stream>>>(w_proj, wproj16, C_ * C_);

  // (16384/32)*(1152/32) = 18432 waves / 8 = 2304 blocks
  qkv_gemm_kernel<<<2304, 256, 0, stream>>>(x16, wqkv16, q16, k16, vt16);

  // 48 (b,h) * 16 blocks (8 strips each) = 768 blocks
  attention_kernel<<<768, 256, 0, stream>>>(q16, k16, vt16, o16, attn);

  // (16384/32)*(384/32) = 6144 waves / 8 = 768 blocks
  proj_gemm_kernel<<<768, 256, 0, stream>>>(o16, wproj16, b_proj, out);
}
